// Head_9783935500842
// MI455X (gfx1250) — matrix operand order repaired and checked by reading, not compiled, not run on hardware
//
#include <hip/hip_runtime.h>

// Problem constants (from reference)
constexpr int BB = 8;
constexpr int TT = 2048;
constexpr int CC = 256;
constexpr int HH = 64;

typedef __attribute__((ext_vector_type(16))) _Float16 v16h;
typedef __attribute__((ext_vector_type(8)))  _Float16 v8h;
typedef __attribute__((ext_vector_type(8)))  float    v8f;
typedef __attribute__((ext_vector_type(4)))  float    v4f;

// ---------------------------------------------------------------------------
// WMMA helper: D = A(16x32 f16) * B(32x16 f16) + C(16x16 f32)
// ---------------------------------------------------------------------------
__device__ __forceinline__ v8f wmma32(v16h a, v16h b, v8f c) {
    return __builtin_amdgcn_wmma_f32_16x16x32_f16(
        /*neg_a=*/false, a, /*neg_b=*/false, b,
        /*c_mod=*/(short)0, c, /*reuse_a=*/false, /*reuse_b=*/false);
}

// A-operand layout load (16-bit A 16x32): per lane, elements p[0..7] and p[16..23].
// Caller passes p already offset by koffA = (lane<16)?0:8.
__device__ __forceinline__ v16h loadA16(const _Float16* p) {
    v8h lo = *(const v8h*)(p);
    v8h hi = *(const v8h*)(p + 16);
    return __builtin_shufflevector(lo, hi, 0,1,2,3,4,5,6,7,8,9,10,11,12,13,14,15);
}

// B-operand layout load (16-bit B 32x16): per lane, elements p[0..7] and p[16..23].
// Caller passes p already offset by koffB = (lane<16)?0:8.
__device__ __forceinline__ v16h loadB16(const _Float16* p) {
    v8h lo = *(const v8h*)(p);
    v8h hi = *(const v8h*)(p + 16);
    return __builtin_shufflevector(lo, hi, 0,1,2,3,4,5,6,7,8,9,10,11,12,13,14,15);
}

// A-operand load with f32 -> f16 conversion (for x).
__device__ __forceinline__ v16h loadA_f32(const float* p) {
    v4f x0 = *(const v4f*)(p);
    v4f x1 = *(const v4f*)(p + 4);
    v4f x2 = *(const v4f*)(p + 16);
    v4f x3 = *(const v4f*)(p + 20);
    v16h a;
#pragma unroll
    for (int i = 0; i < 4; ++i) {
        a[i]      = (_Float16)x0[i];
        a[4 + i]  = (_Float16)x1[i];
        a[8 + i]  = (_Float16)x2[i];
        a[12 + i] = (_Float16)x3[i];
    }
    return a;
}

// ---------------------------------------------------------------------------
// Online-softmax update for one 16x64 score block held as 4 C-layout tiles.
// Per lane: 8 rows (rbase..rbase+7), one column per tile.  Row statistics are
// reduced across the 16-lane halves of the wave (wave32) with shfl_xor and end
// up replicated, so every lane holds correct stats for its 8 rows.
// On exit s[jt][vr] holds p = exp(s - m_new)  (0 for masked entries).
// ---------------------------------------------------------------------------
__device__ __forceinline__ void softmax_update(v8f (&s)[4], float (&mr)[8],
                                               float (&lr)[8], v8f (&oacc)[4]) {
#pragma unroll
    for (int vr = 0; vr < 8; ++vr) {
        float mx = fmaxf(fmaxf(s[0][vr], s[1][vr]), fmaxf(s[2][vr], s[3][vr]));
#pragma unroll
        for (int msk = 1; msk <= 8; msk <<= 1)
            mx = fmaxf(mx, __shfl_xor(mx, msk, 32));
        const float mnew  = fmaxf(mr[vr], mx);
        const float alpha = __expf(mr[vr] - mnew);
        mr[vr] = mnew;
        float rs = 0.0f;
#pragma unroll
        for (int jt = 0; jt < 4; ++jt) {
            const float p = __expf(s[jt][vr] - mnew);
            s[jt][vr] = p;
            rs += p;
        }
#pragma unroll
        for (int msk = 1; msk <= 8; msk <<= 1)
            rs += __shfl_xor(rs, msk, 32);
        lr[vr] = lr[vr] * alpha + rs;
#pragma unroll
        for (int nt = 0; nt < 4; ++nt) oacc[nt][vr] *= alpha;
    }
}

// ---------------------------------------------------------------------------
// Kernel 1: convert weights to f16, transposed: wt16[w][h][c] = (f16) W[c][h]
// w order: 0 -> Wq, 1 -> Wk, 2 -> Wv
// ---------------------------------------------------------------------------
__global__ __launch_bounds__(256) void convert_w_kernel(
    const float* __restrict__ Wq, const float* __restrict__ Wk,
    const float* __restrict__ Wv, _Float16* __restrict__ wt16) {
    const int w = blockIdx.x;
    const float* W = (w == 0) ? Wq : (w == 1) ? Wk : Wv;
    _Float16* dst = wt16 + (size_t)w * HH * CC;
    for (int idx = threadIdx.x; idx < HH * CC; idx += 256) {
        const int h = idx / CC;
        const int c = idx - h * CC;
        dst[idx] = (_Float16)W[c * HH + h];
    }
}

// ---------------------------------------------------------------------------
// Kernel 2: projections.  One wave computes a 16x64 tile of q/k/v.
//   q16, k16 stored [B, T, H] f16;  v stored transposed vT16 [B, H, T] f16.
// grid = (T/16, B, 3), block = 32 (one wave).
// ---------------------------------------------------------------------------
__global__ __launch_bounds__(32) void proj_kernel(
    const float* __restrict__ x, const _Float16* __restrict__ wt16,
    _Float16* __restrict__ q16, _Float16* __restrict__ k16,
    _Float16* __restrict__ vT16) {
    const int lane  = threadIdx.x;
    const int tbase = blockIdx.x * 16;
    const int b     = blockIdx.y;
    const int w     = blockIdx.z;

    const int mrow  = lane & 15;
    const int koffA = (lane < 16) ? 0 : 8;
    const int koffB = (lane < 16) ? 0 : 8;
    const int rbase = (lane < 16) ? 0 : 8;

    const _Float16* wt = wt16 + (size_t)w * HH * CC;   // [h][c]
    const float* xrow  = x + ((size_t)(b * TT + tbase + mrow)) * CC;

    v8f acc[4];
#pragma unroll
    for (int nt = 0; nt < 4; ++nt) acc[nt] = (v8f)(0.0f);

#pragma unroll
    for (int cs = 0; cs < CC / 32; ++cs) {
        const int c0 = cs * 32;
        v16h a = loadA_f32(xrow + c0 + koffA);
#pragma unroll
        for (int nt = 0; nt < 4; ++nt) {
            const _Float16* wp = wt + (size_t)(nt * 16 + mrow) * CC + c0 + koffB;
            acc[nt] = wmma32(a, loadB16(wp), acc[nt]);
        }
    }

    if (w < 2) {
        _Float16* o = (w == 0 ? q16 : k16) + ((size_t)(b * TT + tbase)) * HH;
#pragma unroll
        for (int nt = 0; nt < 4; ++nt)
#pragma unroll
            for (int vr = 0; vr < 8; ++vr)
                o[(size_t)(rbase + vr) * HH + nt * 16 + mrow] = (_Float16)acc[nt][vr];
    } else {
        _Float16* o = vT16 + (size_t)b * HH * TT;
#pragma unroll
        for (int nt = 0; nt < 4; ++nt) {
            v8h pk;
#pragma unroll
            for (int vr = 0; vr < 8; ++vr) pk[vr] = (_Float16)acc[nt][vr];
            *(v8h*)(o + (size_t)(nt * 16 + mrow) * TT + tbase + rbase) = pk;
        }
    }
}

// ---------------------------------------------------------------------------
// Kernel 3: flash attention.  One wave handles 16 query rows; loops over
// 64-column key chunks.  Full (below-diagonal) chunks skip causal masking
// entirely; the single diagonal tail chunk masks per element and skips fully
// dead S-tiles / PV K-steps at uniform-branch granularity.
// grid = (T/16, B), block = 32 (one wave -> barriers are NOPs, LDS in-order).
// ---------------------------------------------------------------------------
__global__ __launch_bounds__(32) void flash_kernel(
    const _Float16* __restrict__ q16, const _Float16* __restrict__ k16,
    const _Float16* __restrict__ vT16, float* __restrict__ out) {
    __shared__ _Float16 plds[16 * 64];

    const int lane  = threadIdx.x;
    const int qbase = blockIdx.x * 16;
    const int b     = blockIdx.y;

    const int mrow  = lane & 15;            // A-layout row / C-layout col
    const int lcol  = lane & 15;
    const int koffA = (lane < 16) ? 0 : 8;
    const int koffB = (lane < 16) ? 0 : 8;
    const int rbase = (lane < 16) ? 0 : 8;  // C-layout row base

    // q A-tiles (held in registers for the whole loop): H=64 -> two K-steps
    const _Float16* qrow = q16 + ((size_t)(b * TT + qbase + mrow)) * HH;
    const v16h aq0 = loadA16(qrow + 0  + koffA);
    const v16h aq1 = loadA16(qrow + 32 + koffA);

    const float scale = 0.125f;  // H^-0.5

    float mr[8], lr[8];
    v8f oacc[4];
#pragma unroll
    for (int vr = 0; vr < 8; ++vr) { mr[vr] = -1e30f; lr[vr] = 0.0f; }
#pragma unroll
    for (int nt = 0; nt < 4; ++nt) oacc[nt] = (v8f)(0.0f);

    const _Float16* kbb = k16  + (size_t)b * TT * HH;
    const _Float16* vbb = vT16 + (size_t)b * HH * TT;

    const int nchunk = (qbase + 16 + 63) >> 6;  // chunks covering causal range
    const int nfull  = qbase >> 6;              // chunks fully below diagonal

    // ---------------- full chunks: no masking needed ----------------
    for (int ch = 0; ch < nfull; ++ch) {
        const int j0 = ch * 64;

        v8f s[4];
#pragma unroll
        for (int jt = 0; jt < 4; ++jt) {
            const _Float16* krow = kbb + (size_t)(j0 + jt * 16 + lcol) * HH;
            v8f z = (v8f)(0.0f);
            z = wmma32(aq0, loadB16(krow + 0  + koffB), z);
            z = wmma32(aq1, loadB16(krow + 32 + koffB), z);
#pragma unroll
            for (int vr = 0; vr < 8; ++vr) z[vr] *= scale;
            s[jt] = z;
        }

        // prefetch next key chunk (global_prefetch_b8)
        __builtin_prefetch(kbb + (size_t)(j0 + 64 + lcol) * HH, 0, 1);

        softmax_update(s, mr, lr, oacc);

        // bounce P through LDS: C-layout -> A-layout
#pragma unroll
        for (int jt = 0; jt < 4; ++jt)
#pragma unroll
            for (int vr = 0; vr < 8; ++vr)
                plds[(rbase + vr) * 64 + jt * 16 + lcol] = (_Float16)s[jt][vr];
        __syncthreads();  // single-wave WG: NOP barrier; orders LDS for compiler

#pragma unroll
        for (int ks = 0; ks < 2; ++ks) {
            v16h ap = loadA16(plds + mrow * 64 + ks * 32 + koffA);
#pragma unroll
            for (int nt = 0; nt < 4; ++nt) {
                const _Float16* vrow =
                    vbb + (size_t)(nt * 16 + lcol) * TT + j0 + ks * 32 + koffB;
                oacc[nt] = wmma32(ap, loadB16(vrow), oacc[nt]);
            }
        }
        __syncthreads();
    }

    // ---------------- diagonal tail chunks: causal masking ----------------
    const int rowmax = qbase + 15;
    for (int ch = nfull; ch < nchunk; ++ch) {
        const int j0 = ch * 64;

        v8f s[4];
#pragma unroll
        for (int jt = 0; jt < 4; ++jt) {
            if (j0 + jt * 16 <= rowmax) {  // tile touches the causal region
                const _Float16* krow = kbb + (size_t)(j0 + jt * 16 + lcol) * HH;
                v8f z = (v8f)(0.0f);
                z = wmma32(aq0, loadB16(krow + 0  + koffB), z);
                z = wmma32(aq1, loadB16(krow + 32 + koffB), z);
                const int col = j0 + jt * 16 + lcol;
#pragma unroll
                for (int vr = 0; vr < 8; ++vr) {
                    const int rowg = qbase + rbase + vr;
                    z[vr] = (col <= rowg) ? z[vr] * scale : -1e30f;
                }
                s[jt] = z;
            } else {                        // fully above diagonal
                s[jt] = (v8f)(-1e30f);
            }
        }

        softmax_update(s, mr, lr, oacc);    // masked entries -> p = 0

#pragma unroll
        for (int jt = 0; jt < 4; ++jt)
#pragma unroll
            for (int vr = 0; vr < 8; ++vr)
                plds[(rbase + vr) * 64 + jt * 16 + lcol] = (_Float16)s[jt][vr];
        __syncthreads();

#pragma unroll
        for (int ks = 0; ks < 2; ++ks) {
            if (j0 + ks * 32 <= rowmax) {   // skip fully-masked PV K-step
                v16h ap = loadA16(plds + mrow * 64 + ks * 32 + koffA);
#pragma unroll
                for (int nt = 0; nt < 4; ++nt) {
                    const _Float16* vrow =
                        vbb + (size_t)(nt * 16 + lcol) * TT + j0 + ks * 32 + koffB;
                    oacc[nt] = wmma32(ap, loadB16(vrow), oacc[nt]);
                }
            }
        }
        __syncthreads();
    }

    // ---- finalize: divide by row sums, store fp32 [B,T,H] ----
    float rinv[8];
#pragma unroll
    for (int vr = 0; vr < 8; ++vr) rinv[vr] = 1.0f / lr[vr];
#pragma unroll
    for (int nt = 0; nt < 4; ++nt)
#pragma unroll
        for (int vr = 0; vr < 8; ++vr)
            out[((size_t)(b * TT + qbase + rbase + vr)) * HH + nt * 16 + lcol] =
                oacc[nt][vr] * rinv[vr];
}

// ---------------------------------------------------------------------------
// Launch.  Inputs: x [B,T,C] f32, Wk [C,H] f32, Wq [C,H] f32, Wv [C,H] f32.
// Output: [B,T,H] f32.  Workspace: q16 | k16 | vT16 | wt16  (~6.2 MB).
// ---------------------------------------------------------------------------
extern "C" void kernel_launch(void* const* d_in, const int* in_sizes, int n_in,
                              void* d_out, int out_size, void* d_ws, size_t ws_size,
                              hipStream_t stream) {
    const float* x  = (const float*)d_in[0];
    const float* Wk = (const float*)d_in[1];
    const float* Wq = (const float*)d_in[2];
    const float* Wv = (const float*)d_in[3];
    float* out = (float*)d_out;

    char* ws = (char*)d_ws;
    constexpr size_t QKV_BYTES = (size_t)BB * TT * HH * sizeof(_Float16);  // 2 MB
    _Float16* q16  = (_Float16*)(ws);
    _Float16* k16  = (_Float16*)(ws + QKV_BYTES);
    _Float16* vT16 = (_Float16*)(ws + 2 * QKV_BYTES);
    _Float16* wt16 = (_Float16*)(ws + 3 * QKV_BYTES);

    convert_w_kernel<<<dim3(3), 256, 0, stream>>>(Wq, Wk, Wv, wt16);
    proj_kernel<<<dim3(TT / 16, BB, 3), 32, 0, stream>>>(x, wt16, q16, k16, vT16);
    flash_kernel<<<dim3(TT / 16, BB), 32, 0, stream>>>(q16, k16, vT16, out);
}
